// QuantumDMRGLayer_71536975282621
// MI455X (gfx1250) — compile-verified, matrix operand order checked by reading, not run on hardware
//
#include <hip/hip_runtime.h>

typedef __attribute__((ext_vector_type(8)))  float  v8f;
typedef __attribute__((ext_vector_type(16))) __bf16 v16bf;

#define DIMVEC 196
#define POS    98
#define NLAB   10
#define MDIM   128
#define NBATCH 8192
#define TB     128          // batch rows per workgroup
#define NTHR   256          // 8 wave32
#define NWAVE  8
#define CPITCH 132          // padded f32 pitch for carry rows (bank-conflict free)
#define LSTEPS 97
#define RSTEPS 96

#define FRAGSTRIDE 528                      // 512 + 16 elems pad: upper half-wave +16B (2-way max conflicts)
#define FRAG_ELEMS (64 * FRAGSTRIDE)        // 2p * 4c * 8t frags per buffer = 33792 bf16
#define CARRY_FLOATS (NWAVE * 16 * CPITCH)  // 16896
#define SMEM_BYTES (2 * FRAG_ELEMS * 2 + CARRY_FLOATS * 4)  // 135168 + 67584 = 202752 B

__device__ __forceinline__ int frag_off(int p, int c, int t) {
    return ((p * 4 + c) * 8 + t) * FRAGSTRIDE;
}
// element slot of fragment-lane fl (0..31): 32B stride, +16B skew for upper half-wave
__device__ __forceinline__ int lslot(int fl) {
    return (fl << 4) + ((fl >> 4) << 3);
}

// Packed f32x2 -> bf16x2 in one v_cvt_pk_bf16_f32 when available.
__device__ __forceinline__ unsigned pk2(float a, float b) {
#if __has_builtin(__builtin_amdgcn_cvt_pk_bf16_f32)
    auto r = __builtin_amdgcn_cvt_pk_bf16_f32(a, b);
    return __builtin_bit_cast(unsigned, r);
#else
    union { __bf16 h[2]; unsigned u; } v;
    v.h[0] = (__bf16)a; v.h[1] = (__bf16)b;
    return v.u;
#endif
}

// ---- Stage B = W[p][K][N] (plain, left scan). Two adjacent K rows per iter -> packed b32 stores.
// Frag element (fl, j): N = 16*t + fl%16 ; K = 32*c + (j<8 ? j : j+8) + (fl>=16 ? 8 : 0)
__device__ __forceinline__ void stage_plain(const float* __restrict__ W, __bf16* frag, int tid) {
    int wv = tid >> 5, ln = tid & 31;
    int n0 = ln << 2;                     // 4 consecutive N per lane -> coalesced float4 rows
    int t = n0 >> 4, lo = n0 & 15;
#pragma unroll
    for (int it = 0; it < 16; ++it) {
        int pr = wv + (it << 3);          // K-row pair index, 0..127
        int r0 = pr << 1;                 // even global row (p,K)
        int p = r0 >> 7, K = r0 & 127;    // K even
        const float4 wa = *(const float4*)(W + (size_t)r0 * MDIM + n0);
        const float4 wb = *(const float4*)(W + (size_t)(r0 + 1) * MDIM + n0);
        int c  = K >> 5, kr = K & 31;
        int hi = (kr >> 3) & 1;
        int j  = ((kr >> 4) << 3) | (kr & 7);       // even; row K+1 -> j+1 (same octet)
        int base = frag_off(p, c, t) + j;
        *(unsigned*)(frag + base + lslot(hi * 16 + lo + 0)) = pk2(wa.x, wb.x);
        *(unsigned*)(frag + base + lslot(hi * 16 + lo + 1)) = pk2(wa.y, wb.y);
        *(unsigned*)(frag + base + lslot(hi * 16 + lo + 2)) = pk2(wa.z, wb.z);
        *(unsigned*)(frag + base + lslot(hi * 16 + lo + 3)) = pk2(wa.w, wb.w);
    }
}

// ---- Stage B = W[p][N][K] (transposed, right scan). 32 lanes tile the 128 K of one (p,N) row;
// 4 contiguous fragment elements per lane -> one packed b64 store.
__device__ __forceinline__ void stage_trans(const float* __restrict__ W, __bf16* frag, int tid) {
    int wv = tid >> 5, ln = tid & 31;
    int c   = ln >> 3;
    int hib = (ln >> 2) & 1;
    int g   = ln & 3;                     // j group: j in {4g..4g+3}
    int j0  = g << 2;
    int Kst = (c << 5) + (hib ? 8 : 0) + ((g < 2) ? (g << 2) : ((g << 2) + 8));
#pragma unroll
    for (int it = 0; it < 32; ++it) {
        int rowIdx = wv + (it << 3);      // (p,N) row
        int p = rowIdx >> 7, N = rowIdx & 127;
        const float4 w = *(const float4*)(W + (size_t)rowIdx * MDIM + Kst);
        int t = N >> 4, lo = N & 15;
        uint2 u;
        u.x = pk2(w.x, w.y);
        u.y = pk2(w.z, w.w);
        *(uint2*)(frag + frag_off(p, c, t) + lslot(hib * 16 + lo) + j0) = u;
    }
}

// ---- Stage B[K=m][N=n] = w_label[p][m][n][l] (N-stride = NLAB floats). Paired rows like plain.
__device__ __forceinline__ void stage_label(const float* __restrict__ WL, __bf16* frag, int tid, int l) {
    int wv = tid >> 5, ln = tid & 31;
    int n0 = ln << 2;
    int t = n0 >> 4, lo = n0 & 15;
#pragma unroll
    for (int it = 0; it < 16; ++it) {
        int pr = wv + (it << 3);
        int r0 = pr << 1;
        int p = r0 >> 7, K = r0 & 127;
        const float* g0 = WL + (size_t)(r0 * MDIM + n0) * NLAB + l;
        const float* g1 = g0 + (size_t)MDIM * NLAB;
        float a0 = g0[0], a1 = g0[NLAB], a2 = g0[2 * NLAB], a3 = g0[3 * NLAB];
        float b0 = g1[0], b1 = g1[NLAB], b2 = g1[2 * NLAB], b3 = g1[3 * NLAB];
        int c  = K >> 5, kr = K & 31;
        int hi = (kr >> 3) & 1;
        int j  = ((kr >> 4) << 3) | (kr & 7);
        int base = frag_off(p, c, t) + j;
        *(unsigned*)(frag + base + lslot(hi * 16 + lo + 0)) = pk2(a0, b0);
        *(unsigned*)(frag + base + lslot(hi * 16 + lo + 1)) = pk2(a1, b1);
        *(unsigned*)(frag + base + lslot(hi * 16 + lo + 2)) = pk2(a2, b2);
        *(unsigned*)(frag + base + lslot(hi * 16 + lo + 3)) = pk2(a3, b3);
    }
}

// ---- Build bf16 A-fragments (two scaled copies) from this wave's f32 carry rows in LDS.
// Built as 8 packed u32 words per chunk: 2 muls + 1 v_cvt_pk per bf16 pair.
__device__ __forceinline__ void build_A(const float* cw, float sx0, float sx1,
                                        int myM, int hi, v16bf* A0, v16bf* A1) {
#pragma unroll
    for (int c = 0; c < 4; ++c) {
        int k0 = (c << 5) + (hi ? 8 : 0);
        const float* src = cw + myM * CPITCH + k0;
        float4 u0 = *(const float4*)(src);
        float4 u1 = *(const float4*)(src + 4);
        float4 u2 = *(const float4*)(src + 16);
        float4 u3 = *(const float4*)(src + 20);
        float v[16] = {u0.x, u0.y, u0.z, u0.w, u1.x, u1.y, u1.z, u1.w,
                       u2.x, u2.y, u2.z, u2.w, u3.x, u3.y, u3.z, u3.w};
        union { unsigned u[8]; v16bf b; } a0, a1;
#pragma unroll
        for (int q = 0; q < 8; ++q) {
            a0.u[q] = pk2(sx0 * v[2 * q], sx0 * v[2 * q + 1]);
            a1.u[q] = pk2(sx1 * v[2 * q], sx1 * v[2 * q + 1]);
        }
        A0[c] = a0.b;
        A1[c] = a1.b;
    }
}

// ---- One 16x16 output tile: acc = A0 @ W0 + A1 @ W1  (8 WMMAs, f32 accumulate)
__device__ __forceinline__ v8f tile_mma(const __bf16* frag, const v16bf* A0, const v16bf* A1,
                                        int t, int laneOff) {
    v8f acc = {0.f, 0.f, 0.f, 0.f, 0.f, 0.f, 0.f, 0.f};
#pragma unroll
    for (int c = 0; c < 4; ++c) {
        v16bf B0 = *(const v16bf*)(frag + frag_off(0, c, t) + laneOff);
        acc = __builtin_amdgcn_wmma_f32_16x16x32_bf16(false, A0[c], false, B0,
                                                      (short)0, acc, false, false);
        v16bf B1 = *(const v16bf*)(frag + frag_off(1, c, t) + laneOff);
        acc = __builtin_amdgcn_wmma_f32_16x16x32_bf16(false, A1[c], false, B1,
                                                      (short)0, acc, false, false);
    }
    return acc;
}

// ================= Scan kernel (DIR=0: left, DIR=1: right/reversed/transposed) ===============
template <int DIR>
__global__ __launch_bounds__(NTHR) void scan_kernel(const float* __restrict__ x,
                                                    const float* __restrict__ wInit,
                                                    const float* __restrict__ wScan,
                                                    float* __restrict__ wsOut) {
    extern __shared__ char smem[];
    __bf16* fragB = (__bf16*)smem;                       // two buffers of FRAG_ELEMS
    float*  carry = (float*)(smem + 2 * FRAG_ELEMS * 2);

    const int tid  = threadIdx.x;
    const int wave = tid >> 5, lane = tid & 31;
    const int myM  = lane & 15, hi = lane >> 4;
    const int laneOff = lslot(lane);
    const int row0 = blockIdx.x * TB;
    float* cw = carry + wave * (16 * CPITCH);

    // carry init: x[:,i0,:] @ wInit
    {
        const int i0 = DIR ? (DIMVEC - 1) : 0;
        for (int idx = tid; idx < TB * MDIM; idx += NTHR) {
            int r = idx >> 7, n = idx & 127;
            const float2 xv = *(const float2*)(x + ((size_t)(row0 + r) * DIMVEC + i0) * 2);
            carry[(r >> 4) * (16 * CPITCH) + (r & 15) * CPITCH + n] =
                xv.x * wInit[n] + xv.y * wInit[MDIM + n];
        }
    }

    const int NS    = DIR ? RSTEPS : LSTEPS;
    const int myrow = row0 + wave * 16 + myM;

    // pipeline prologue: stage step 0 into buffer 0
    {
        const int s0 = DIR ? (NS - 1) : 0;
        const float* W0p = wScan + (size_t)s0 * 2 * MDIM * MDIM;
        if (DIR) stage_trans(W0p, fragB, tid);
        else     stage_plain(W0p, fragB, tid);
    }
    __syncthreads();

    for (int ss = 0; ss < NS; ++ss) {
        const int s = DIR ? (NS - 1 - ss) : ss;
        const __bf16* cur = fragB + (ss & 1) * FRAG_ELEMS;
        __bf16*       nxt = fragB + ((ss + 1) & 1) * FRAG_ELEMS;

        // stage next step into the other buffer (overlaps with this step's WMMAs via wave skew)
        if (ss + 1 < NS) {
            const int sn = DIR ? (s - 1) : (s + 1);
            const float* Wn = wScan + (size_t)sn * 2 * MDIM * MDIM;
            if (DIR) stage_trans(Wn, nxt, tid);
            else     stage_plain(Wn, nxt, tid);
            if (ss + 2 < NS) {
                const int sp = DIR ? (s - 2) : (s + 2);
                __builtin_prefetch(wScan + (size_t)sp * 2 * MDIM * MDIM + tid * 128, 0, 1);
            }
        }

        const int i = DIR ? (POS + 1 + s) : (1 + s);
        const float2 xv = *(const float2*)(x + ((size_t)myrow * DIMVEC + i) * 2);
        v16bf A0[4], A1[4];
        build_A(cw, xv.x, xv.y, myM, hi, A0, A1);   // A fully in regs -> carry may be overwritten

#pragma unroll
        for (int t = 0; t < 8; ++t) {
            v8f acc = tile_mma(cur, A0, A1, t, laneOff);
#pragma unroll
            for (int r = 0; r < 8; ++r)             // D layout: VGPR r -> row r (+8 for hi lanes)
                cw[(r + hi * 8) * CPITCH + t * 16 + myM] = acc[r];
        }
        __syncthreads();   // all waves done reading `cur` and writing `nxt`
    }

    float* dst = wsOut + (size_t)row0 * MDIM;
    for (int idx = tid; idx < TB * MDIM / 4; idx += NTHR) {
        int r = idx >> 5, q = idx & 31;
        *(float4*)(dst + (size_t)r * MDIM + q * 4) =
            *(const float4*)(carry + (r >> 4) * (16 * CPITCH) + (r & 15) * CPITCH + q * 4);
    }
}

// ================= Final label contraction =================
__global__ __launch_bounds__(NTHR) void label_kernel(const float* __restrict__ x,
                                                     const float* __restrict__ wLab,
                                                     const float* __restrict__ leftWS,
                                                     const float* __restrict__ rightWS,
                                                     float* __restrict__ out) {
    extern __shared__ char smem[];
    __bf16* fragB = (__bf16*)smem;
    float*  carry = (float*)(smem + 2 * FRAG_ELEMS * 2);

    const int tid  = threadIdx.x;
    const int wave = tid >> 5, lane = tid & 31;
    const int myM  = lane & 15, hi = lane >> 4;
    const int laneOff = lslot(lane);
    const int row0 = blockIdx.x * TB;
    const float* cw = carry + wave * (16 * CPITCH);

    for (int idx = tid; idx < TB * MDIM; idx += NTHR) {
        int r = idx >> 7, n = idx & 127;
        carry[(r >> 4) * (16 * CPITCH) + (r & 15) * CPITCH + n] =
            leftWS[(size_t)(row0 + r) * MDIM + n];
    }
    stage_label(wLab, fragB, tid, 0);
    __syncthreads();

    const int myrow = row0 + wave * 16 + myM;
    const float2 xv = *(const float2*)(x + ((size_t)myrow * DIMVEC + POS) * 2);
    v16bf A0[4], A1[4];
    build_A(cw, xv.x, xv.y, myM, hi, A0, A1);       // A is invariant across labels

    float rv[64];                                    // right[b,n] values matching D layout
#pragma unroll
    for (int t = 0; t < 8; ++t)
#pragma unroll
        for (int r = 0; r < 8; ++r)
            rv[t * 8 + r] = rightWS[(size_t)(row0 + wave * 16 + hi * 8 + r) * MDIM + t * 16 + myM];

    for (int l = 0; l < NLAB; ++l) {
        const __bf16* cur = fragB + (l & 1) * FRAG_ELEMS;
        __bf16*       nxt = fragB + ((l + 1) & 1) * FRAG_ELEMS;
        if (l + 1 < NLAB) stage_label(wLab, nxt, tid, l + 1);

        float sums[8] = {0, 0, 0, 0, 0, 0, 0, 0};
#pragma unroll
        for (int t = 0; t < 8; ++t) {
            v8f acc = tile_mma(cur, A0, A1, t, laneOff);
#pragma unroll
            for (int r = 0; r < 8; ++r)
                sums[r] += acc[r] * rv[t * 8 + r];
        }
#pragma unroll
        for (int r = 0; r < 8; ++r) {
#pragma unroll
            for (int m = 1; m < 16; m <<= 1)
                sums[r] += __shfl_xor(sums[r], m, 32);  // reduce over the 16 N-lanes
        }
        if (myM == 0) {
#pragma unroll
            for (int r = 0; r < 8; ++r)
                out[(size_t)(row0 + wave * 16 + hi * 8 + r) * NLAB + l] = sums[r];
        }
        __syncthreads();
    }
}

extern "C" void kernel_launch(void* const* d_in, const int* in_sizes, int n_in,
                              void* d_out, int out_size, void* d_ws, size_t ws_size,
                              hipStream_t stream) {
    const float* x      = (const float*)d_in[0];
    const float* w0     = (const float*)d_in[1];
    const float* wleft  = (const float*)d_in[2];
    const float* wlabel = (const float*)d_in[3];
    const float* wright = (const float*)d_in[4];
    const float* wend   = (const float*)d_in[5];
    float* out = (float*)d_out;

    float* lws = (float*)d_ws;                 // left  carries: 8192*128 f32
    float* rws = lws + (size_t)NBATCH * MDIM;  // right carries: 8192*128 f32

    (void)hipFuncSetAttribute((const void*)scan_kernel<0>,
                              hipFuncAttributeMaxDynamicSharedMemorySize, SMEM_BYTES);
    (void)hipFuncSetAttribute((const void*)scan_kernel<1>,
                              hipFuncAttributeMaxDynamicSharedMemorySize, SMEM_BYTES);
    (void)hipFuncSetAttribute((const void*)label_kernel,
                              hipFuncAttributeMaxDynamicSharedMemorySize, SMEM_BYTES);

    dim3 grid(NBATCH / TB), blk(NTHR);
    scan_kernel<0><<<grid, blk, SMEM_BYTES, stream>>>(x, w0,   wleft,  lws);
    scan_kernel<1><<<grid, blk, SMEM_BYTES, stream>>>(x, wend, wright, rws);
    label_kernel<<<grid, blk, SMEM_BYTES, stream>>>(x, wlabel, lws, rws, out);
}
